// SparseAttention_13846974562912
// MI455X (gfx1250) — matrix operand order repaired and checked by reading, not compiled, not run on hardware
//
#include <hip/hip_runtime.h>
#include <hip/hip_bf16.h>

// ---------------------------------------------------------------------------
// Problem constants (from reference): B=8, N=8192, C=512, heads=8, dh=64,
// BLOCK_SIZE=32 -> nb=256 blocks, WINDOW=64 blocks, scale = dh^-0.5 = 0.125.
// Attention contracts over FULL C=512 (heads only permute QKV columns).
// ---------------------------------------------------------------------------
#define TOK   65536      // B*N tokens
#define CDIM  512
#define NQKV  1536
#define NBLK  256        // blocks per sequence
#define BS    32         // tokens per block
#define WIN   64         // window radius (blocks)
#define QLD   520        // LDS row stride (ushorts) for 512-wide rows (16B aligned, bank-spread)
#define TLD   40         // LDS row stride (ushorts) for 32-wide tiles  (16B aligned, bank-spread)

typedef __bf16 bf16_t;
typedef __attribute__((ext_vector_type(16))) __bf16 v16bf;
typedef __attribute__((ext_vector_type(2)))  __bf16 v2bf;
typedef __attribute__((ext_vector_type(8)))  float  v8f;

union FragAB {
    v16bf bf;
    uint4 q[2];
    unsigned short u[16];
};

// float -> bf16 via hardware convert (v_cvt), not software RNE bit math
__device__ __forceinline__ unsigned short f2bf(float x) {
    __bf16 h = (__bf16)x;
    return __builtin_bit_cast(unsigned short, h);
}
// two floats -> packed bf16x2 (v_cvt_pk_bf16_f32), no half-reshuffling
__device__ __forceinline__ unsigned int pack2bf(float a, float b) {
    v2bf t;
    t.x = (__bf16)a;
    t.y = (__bf16)b;
    return __builtin_bit_cast(unsigned int, t);
}

// A fragment (16x32 bf16, row-major LDS [rows][ldm]):
// lane (r,half): e in 0..7 -> K = k0 + 8*half + e ; e in 8..15 -> K = k0 + 16 + 8*half + (e-8)
__device__ __forceinline__ FragAB load_frag_a(const unsigned short* lds, int row,
                                              int ldm, int k0, int r, int half) {
    FragAB f;
    const unsigned short* p = lds + (row + r) * ldm + k0 + 8 * half;
    f.q[0] = *(const uint4*)p;
    f.q[1] = *(const uint4*)(p + 16);
    return f;
}
// B fragment (32x16 bf16) from K-contiguous transposed LDS tile Bt[n][k]:
// lane (r,half): e in 0..7 -> K = k0 + 8*half + e ; e in 8..15 -> K = k0 + 16 + 8*half + (e-8), N = n0 + r
__device__ __forceinline__ FragAB load_frag_bt(const unsigned short* lds, int n0,
                                               int ldn, int k0, int r, int half) {
    FragAB f;
    const unsigned short* p = lds + (n0 + r) * ldn + k0 + 8 * half;
    f.q[0] = *(const uint4*)p;
    f.q[1] = *(const uint4*)(p + 16);
    return f;
}
__device__ __forceinline__ v8f wmma_bf16(FragAB a, FragAB b, v8f c) {
    return __builtin_amdgcn_wmma_f32_16x16x32_bf16(false, a.bf, false, b.bf,
                                                   (short)0, c, false, false);
}

// ---------------------------------------------------------------------------
// Kernel 1: weight prep. Wqkv_t[1536][512] bf16 with head-interleave folded in,
// permuted bias[1536], Wproj_t[512][512] bf16 (transposed).
// q cols: (oc/64)*192 + 0*64 + oc%64 ; k: +64 ; v: +128
// ---------------------------------------------------------------------------
__global__ void prep_weights(const float* __restrict__ Wqkv, const float* __restrict__ bqkv,
                             const float* __restrict__ Wproj,
                             unsigned short* __restrict__ Wt, float* __restrict__ bperm,
                             unsigned short* __restrict__ Wpt) {
    const int t0 = blockIdx.x * blockDim.x + threadIdx.x;
    const int nthr = gridDim.x * blockDim.x;
    for (int idx = t0; idx < NQKV * CDIM; idx += nthr) {
        int m = idx >> 9, c = idx & 511;
        int part = m >> 9, oc = m & 511;
        int src = (oc >> 6) * 192 + part * 64 + (oc & 63);
        Wt[idx] = f2bf(Wqkv[(size_t)c * NQKV + src]);
    }
    for (int idx = t0; idx < NQKV; idx += nthr) {
        int part = idx >> 9, oc = idx & 511;
        int src = (oc >> 6) * 192 + part * 64 + (oc & 63);
        bperm[idx] = bqkv[src];
    }
    for (int idx = t0; idx < CDIM * CDIM; idx += nthr) {
        int m = idx >> 9, c = idx & 511;
        Wpt[idx] = f2bf(Wproj[(size_t)c * CDIM + m]);
    }
}

// ---------------------------------------------------------------------------
// Kernel 2: QKV GEMM.  out[65536 x 1536] = x(f32) @ Wt^T + bperm, split into
// Q/K/V bf16 buffers [token][512]. Block tile 128x128, BK=32, 8 waves (2x4),
// each wave 64x32 = 8 wmma tiles.
// ---------------------------------------------------------------------------
__global__ __launch_bounds__(256) void qkv_gemm(
    const float* __restrict__ x, const unsigned short* __restrict__ Wt,
    const float* __restrict__ bperm,
    unsigned short* __restrict__ Q, unsigned short* __restrict__ K,
    unsigned short* __restrict__ V) {
    __shared__ unsigned short lds_a[128][TLD];
    __shared__ unsigned short lds_b[128][TLD];
    const int tid = threadIdx.x;
    const int w = tid >> 5, lane = tid & 31, r = lane & 15, half = lane >> 4;
    const int wm = w >> 2, wn = w & 3;
    const int row0 = blockIdx.y * 128;
    const int n0 = blockIdx.x * 128;

    v8f acc[4][2] = {};

    for (int kb = 0; kb < CDIM; kb += 32) {
        for (int it = 0; it < 4; ++it) {                 // A: 128x32 f32 -> bf16
            int e = (tid + it * 256) * 4;
            int ar = e >> 5, ac = e & 31;
            const float* src = x + (size_t)(row0 + ar) * CDIM + kb + ac;
            if (kb + 32 < CDIM) __builtin_prefetch(src + 32, 0, 3);
            float4 v = *(const float4*)src;
            uint2 pk;
            pk.x = pack2bf(v.x, v.y);
            pk.y = pack2bf(v.z, v.w);
            *(uint2*)&lds_a[ar][ac] = pk;
        }
        for (int it = 0; it < 2; ++it) {                 // B: Wt rows n0..n0+127
            int chunk = tid + it * 256;
            int br = chunk >> 2, bc = (chunk & 3) * 8;
            const unsigned short* src = Wt + (size_t)(n0 + br) * CDIM + kb + bc;
            if (kb + 32 < CDIM) __builtin_prefetch(src + 32, 0, 3);
            *(uint4*)&lds_b[br][bc] = *(const uint4*)src;
        }
        __syncthreads();
        for (int mi = 0; mi < 4; ++mi) {
            FragAB a = load_frag_a(&lds_a[0][0], wm * 64 + mi * 16, TLD, 0, r, half);
            for (int ni = 0; ni < 2; ++ni) {
                FragAB b = load_frag_bt(&lds_b[0][0], wn * 32 + ni * 16, TLD, 0, r, half);
                acc[mi][ni] = wmma_bf16(a, b, acc[mi][ni]);
            }
        }
        __syncthreads();
    }
    const int part = n0 >> 9;                       // 512 % 128 == 0 -> uniform per block
    unsigned short* outb = (part == 0) ? Q : (part == 1) ? K : V;
    for (int mi = 0; mi < 4; ++mi)
        for (int ni = 0; ni < 2; ++ni) {
            int ncol = n0 + wn * 32 + ni * 16 + r;
            int ocol = ncol & 511;
            float bias = bperm[ncol];
            for (int i = 0; i < 8; ++i) {
                int mrow = row0 + wm * 64 + mi * 16 + 8 * half + i;
                outb[(size_t)mrow * CDIM + ocol] = f2bf(acc[mi][ni][i] + bias);
            }
        }
}

// ---------------------------------------------------------------------------
// Kernel 3: banded scores + softmax.  One WG per (b, s, 64-row i-tile).
// Whole band (<=192 cols) of S kept in LDS, fused mask+softmax, writes dense
// P[256 problems][256][256] bf16 (zeros outside band). Dynamic LDS ~150KB.
// ---------------------------------------------------------------------------
__global__ __launch_bounds__(256) void attn_scores(
    const unsigned short* __restrict__ Q, const unsigned short* __restrict__ K,
    unsigned short* __restrict__ P) {
    extern __shared__ char smem[];
    unsigned short* qs = (unsigned short*)smem;          // [64][QLD]
    unsigned short* ks = qs + 64 * QLD;                  // [32][QLD]
    float* S    = (float*)(ks + 32 * QLD);               // [64][192]
    float* rmax = S + 64 * 192;                          // [64]
    float* rinv = rmax + 64;                             // [64]

    const int tid = threadIdx.x;
    const int w = tid >> 5, lane = tid & 31, r = lane & 15, half = lane >> 4;
    const int wm = w & 3, wn = w >> 2;
    const int i0 = blockIdx.x * 64, s = blockIdx.y, b = blockIdx.z;
    const int jlo = max(0, i0 - WIN);
    const int jhi = min(NBLK - 1, i0 + 63 + WIN);
    const int width = jhi - jlo + 1;                     // 128 or 192 (mult of 32)
    const size_t base = ((size_t)b * 8192 + s) * CDIM;   // token = b*8192 + blk*32 + s

    for (int it = 0; it < 16; ++it) {                    // Q: 64 rows x 512
        int chunk = tid + it * 256;
        int qr = chunk >> 6, qc = (chunk & 63) * 8;
        *(uint4*)&qs[qr * QLD + qc] =
            *(const uint4*)(Q + base + (size_t)(i0 + qr) * BS * CDIM + qc);
    }
    for (int j0 = jlo; j0 <= jhi; j0 += 32) {
        __syncthreads();
        for (int it = 0; it < 8; ++it) {                 // K: 32 rows x 512
            int chunk = tid + it * 256;
            int kr = chunk >> 6, kc = (chunk & 63) * 8;
            *(uint4*)&ks[kr * QLD + kc] =
                *(const uint4*)(K + base + (size_t)(j0 + kr) * BS * CDIM + kc);
        }
        __syncthreads();
        v8f acc = {};
        for (int kb = 0; kb < CDIM; kb += 32) {          // S = Q . K^T (Bt = K rows)
            FragAB a  = load_frag_a(qs, wm * 16, QLD, kb, r, half);
            FragAB bq = load_frag_bt(ks, wn * 16, QLD, kb, r, half);
            acc = wmma_bf16(a, bq, acc);
        }
        int gj = j0 + wn * 16 + r;
        for (int i = 0; i < 8; ++i) {
            int li = wm * 16 + 8 * half + i;
            int gi = i0 + li;
            bool ok = (gj >= gi - WIN) && (gj <= gi + WIN);
            S[li * 192 + (gj - jlo)] = ok ? acc[i] * 0.125f : -3.0e38f;
        }
    }
    __syncthreads();
    if (tid < 64) {                                      // per-row softmax stats
        float m = -3.4e38f;
        for (int c = 0; c < width; ++c) m = fmaxf(m, S[tid * 192 + c]);
        float sum = 0.f;
        for (int c = 0; c < width; ++c) sum += __expf(S[tid * 192 + c] - m);
        rmax[tid] = m;
        rinv[tid] = 1.0f / sum;
    }
    __syncthreads();
    const size_t pbase = ((size_t)(b * 32 + s) * NBLK + i0) * NBLK;
    for (int idx = tid; idx < 64 * NBLK; idx += 256) {
        int li = idx >> 8, j = idx & 255;
        float v = 0.f;
        if (j >= jlo && j <= jhi)
            v = __expf(S[li * 192 + (j - jlo)] - rmax[li]) * rinv[li];
        P[pbase + (size_t)li * NBLK + j] = f2bf(v);
    }
}

// ---------------------------------------------------------------------------
// Kernel 4: O = P . V over the band's j-tiles. One WG per (b,s,i-tile).
// 8 waves: 4 row-slabs x 2 col-halves; each wave 16x256 = 16 wmma tiles.
// V transposed into LDS so B fragments are K-contiguous.
// ---------------------------------------------------------------------------
__global__ __launch_bounds__(256) void attn_pv(
    const unsigned short* __restrict__ P, const unsigned short* __restrict__ V,
    unsigned short* __restrict__ AO) {
    __shared__ unsigned short pt[64][TLD];
    __shared__ unsigned short vt[512][TLD];
    const int tid = threadIdx.x;
    const int w = tid >> 5, lane = tid & 31, r = lane & 15, half = lane >> 4;
    const int wm = w & 3, wn = w >> 2;
    const int i0 = blockIdx.x * 64, s = blockIdx.y, b = blockIdx.z;
    const int jlo = max(0, i0 - WIN);
    const int jhi = min(NBLK - 1, i0 + 63 + WIN);
    const size_t base  = ((size_t)b * 8192 + s) * CDIM;
    const size_t pbase = ((size_t)(b * 32 + s) * NBLK + i0) * NBLK;

    v8f acc[16] = {};

    for (int j0 = jlo; j0 <= jhi; j0 += 32) {
        __syncthreads();
        {                                                // P tile 64x32
            int pr = tid >> 2, pc = (tid & 3) * 8;
            *(uint4*)&pt[pr][pc] =
                *(const uint4*)(P + pbase + (size_t)pr * NBLK + j0 + pc);
        }
        for (int it = 0; it < 8; ++it) {                 // V 32x512 -> vt[n][k]
            int chunk = tid + it * 256;
            int vr = chunk >> 6, vc = (chunk & 63) * 8;
            uint4 d = *(const uint4*)(V + base + (size_t)(j0 + vr) * BS * CDIM + vc);
            const unsigned short* ds = (const unsigned short*)&d;
            for (int e = 0; e < 8; ++e) vt[vc + e][vr] = ds[e];
        }
        __syncthreads();
        FragAB a = load_frag_a(&pt[0][0], wm * 16, TLD, 0, r, half);
        for (int ni = 0; ni < 16; ++ni) {
            FragAB bv = load_frag_bt(&vt[0][0], wn * 256 + ni * 16, TLD, 0, r, half);
            acc[ni] = wmma_bf16(a, bv, acc[ni]);
        }
    }
    for (int ni = 0; ni < 16; ++ni) {
        int ncol = wn * 256 + ni * 16 + r;
        for (int i = 0; i < 8; ++i) {
            int gi = i0 + wm * 16 + 8 * half + i;
            size_t token = (size_t)b * 8192 + (size_t)gi * BS + s;
            AO[token * CDIM + ncol] = f2bf(acc[ni][i]);
        }
    }
}

// ---------------------------------------------------------------------------
// Kernel 5: output projection.  out(f32) = AO(bf16) @ Wproj_t^T + b_proj.
// ---------------------------------------------------------------------------
__global__ __launch_bounds__(256) void proj_gemm(
    const unsigned short* __restrict__ A, const unsigned short* __restrict__ Wt,
    const float* __restrict__ bias, float* __restrict__ out) {
    __shared__ unsigned short lds_a[128][TLD];
    __shared__ unsigned short lds_b[128][TLD];
    const int tid = threadIdx.x;
    const int w = tid >> 5, lane = tid & 31, r = lane & 15, half = lane >> 4;
    const int wm = w >> 2, wn = w & 3;
    const int row0 = blockIdx.y * 128;
    const int n0 = blockIdx.x * 128;

    v8f acc[4][2] = {};

    for (int kb = 0; kb < CDIM; kb += 32) {
        for (int it = 0; it < 2; ++it) {
            int chunk = tid + it * 256;
            int ar = chunk >> 2, ac = (chunk & 3) * 8;
            const unsigned short* src = A + (size_t)(row0 + ar) * CDIM + kb + ac;
            if (kb + 32 < CDIM) __builtin_prefetch(src + 32, 0, 3);
            *(uint4*)&lds_a[ar][ac] = *(const uint4*)src;
        }
        for (int it = 0; it < 2; ++it) {
            int chunk = tid + it * 256;
            int br = chunk >> 2, bc = (chunk & 3) * 8;
            const unsigned short* src = Wt + (size_t)(n0 + br) * CDIM + kb + bc;
            if (kb + 32 < CDIM) __builtin_prefetch(src + 32, 0, 3);
            *(uint4*)&lds_b[br][bc] = *(const uint4*)src;
        }
        __syncthreads();
        for (int mi = 0; mi < 4; ++mi) {
            FragAB a = load_frag_a(&lds_a[0][0], wm * 64 + mi * 16, TLD, 0, r, half);
            for (int ni = 0; ni < 2; ++ni) {
                FragAB b = load_frag_bt(&lds_b[0][0], wn * 32 + ni * 16, TLD, 0, r, half);
                acc[mi][ni] = wmma_bf16(a, b, acc[mi][ni]);
            }
        }
        __syncthreads();
    }
    for (int mi = 0; mi < 4; ++mi)
        for (int ni = 0; ni < 2; ++ni) {
            int ncol = n0 + wn * 32 + ni * 16 + r;
            float bv = bias[ncol];
            for (int i = 0; i < 8; ++i) {
                int mrow = row0 + wm * 64 + mi * 16 + 8 * half + i;
                out[(size_t)mrow * CDIM + ncol] = acc[mi][ni][i] + bv;
            }
        }
}

// ---------------------------------------------------------------------------
extern "C" void kernel_launch(void* const* d_in, const int* in_sizes, int n_in,
                              void* d_out, int out_size, void* d_ws, size_t ws_size,
                              hipStream_t stream) {
    const float* x     = (const float*)d_in[0];
    const float* Wqkv  = (const float*)d_in[1];
    const float* bqkv  = (const float*)d_in[2];
    const float* Wproj = (const float*)d_in[3];
    const float* bproj = (const float*)d_in[4];
    float* out = (float*)d_out;

    char* ws = (char*)d_ws;
    size_t off = 0;
    auto alloc = [&](size_t bytes) {
        char* p = ws + off;
        off += (bytes + 255) & ~(size_t)255;
        return p;
    };
    unsigned short* Wt    = (unsigned short*)alloc((size_t)NQKV * CDIM * 2);
    float*          bperm = (float*)alloc((size_t)NQKV * 4);
    unsigned short* Wpt   = (unsigned short*)alloc((size_t)CDIM * CDIM * 2);
    unsigned short* Qb    = (unsigned short*)alloc((size_t)TOK * CDIM * 2);
    unsigned short* Kb    = (unsigned short*)alloc((size_t)TOK * CDIM * 2);
    unsigned short* Vb    = (unsigned short*)alloc((size_t)TOK * CDIM * 2);
    unsigned short* Pb    = (unsigned short*)alloc((size_t)256 * NBLK * NBLK * 2);
    unsigned short* AO    = Kb;   // K is dead after attn_scores; reuse its region

    prep_weights<<<dim3(512), 256, 0, stream>>>(Wqkv, bqkv, Wproj, Wt, bperm, Wpt);
    qkv_gemm<<<dim3(NQKV / 128, TOK / 128), 256, 0, stream>>>(x, Wt, bperm, Qb, Kb, Vb);

    size_t dynLds = (size_t)(64 * QLD + 32 * QLD) * 2 + (size_t)64 * 192 * 4 + 2 * 64 * 4;
    hipFuncSetAttribute((const void*)attn_scores,
                        hipFuncAttributeMaxDynamicSharedMemorySize, (int)dynLds);
    attn_scores<<<dim3(4, 32, 8), 256, dynLds, stream>>>(Qb, Kb, Pb);
    attn_pv<<<dim3(4, 32, 8), 256, 0, stream>>>(Pb, Vb, AO);
    proj_gemm<<<dim3(CDIM / 128, TOK / 128), 256, 0, stream>>>(AO, Wpt, bproj, out);
}
